// SharedBackbone_61400852463655
// MI455X (gfx1250) — matrix operand order repaired and checked by reading, not compiled, not run on hardware
//
#include <hip/hip_runtime.h>
#include <hip/hip_bf16.h>

typedef __attribute__((ext_vector_type(16))) _Float16 v16h;
typedef __attribute__((ext_vector_type(8)))  _Float16 v8h;
typedef __attribute__((ext_vector_type(8)))  float    v8f;

#define NN   100000
#define EE   1600000
#define ET   (EE + NN)
#define CH   128
#define NHEAD 4

// ---------------- utility atomics ----------------
__device__ inline void atomicMaxF(float* addr, float value) {
  if (value >= 0.f)
    atomicMax((int*)addr, __float_as_int(value));
  else
    atomicMin((unsigned int*)addr, __float_as_uint(value));
}

// ---------------- prep kernels ----------------
__global__ void cvt_f32_to_f16_kernel(const float* __restrict__ in,
                                      _Float16* __restrict__ out, int n) {
  int i = blockIdx.x * blockDim.x + threadIdx.x;
  if (i < n) out[i] = (_Float16)in[i];
}

// Wt[c][k] = (f16) W[k][c], W is 128x128 row-major
__global__ void transpose_w_kernel(const float* __restrict__ W,
                                   _Float16* __restrict__ Wt) {
  int i = blockIdx.x * blockDim.x + threadIdx.x;   // i = c*128 + k
  if (i >= CH * CH) return;
  int c = i >> 7, k = i & 127;
  Wt[i] = (_Float16)W[k * CH + c];
}

// ---------------- WMMA GEMM:  C[M,128] = A[M,128] * B[128,128] ----------------
// A: f16 row-major (stride 128). Bt: f16, Bt[col][k] row-major (stride 128).
// One 16x16 output tile per wave, 8 waves per block. K = 128 -> 4 WMMA steps.
__global__ void wmma_gemm128_kernel(const _Float16* __restrict__ A,
                                    const _Float16* __restrict__ Bt,
                                    float* __restrict__ C, int M) {
  const int wave = threadIdx.x >> 5;
  const int lane = threadIdx.x & 31;
  const long tile = (long)blockIdx.x * 8 + wave;
  const int colTiles = CH / 16;                       // 8
  const long rowTile = tile / colTiles;
  const int  colTile = (int)(tile % colTiles);
  const int  m0 = (int)rowTile * 16;
  if (m0 >= M) return;
  const int n0 = colTile * 16;

  const int half = lane >> 4;     // 0: lanes 0-15, 1: lanes 16-31
  const int l15  = lane & 15;

  const _Float16* arow = A  + (long)(m0 + l15) * CH;  // A row for this lane
  const _Float16* brow = Bt + (long)(n0 + l15) * CH;  // B column (row of Bt)

  v8f acc = {};
#pragma unroll
  for (int kb = 0; kb < CH; kb += 32) {
    // A fragment (16-bit 16x32 layout):
    //   lanes 0-15 : K = kb+0..7  and kb+16..23
    //   lanes 16-31: K = kb+8..15 and kb+24..31
    const _Float16* ap = arow + kb + half * 8;
    v8h alo = *(const v8h*)(ap);
    v8h ahi = *(const v8h*)(ap + 16);
    v16h afrag = __builtin_shufflevector(alo, ahi,
        0,1,2,3,4,5,6,7,8,9,10,11,12,13,14,15);
    // B fragment (32x16): lanes 0-15 hold K = kb+0..7 and kb+16..23, lanes 16-31 K = kb+8..15 and kb+24..31
    const _Float16* bp = brow + kb + half * 8;
    v8h blo = *(const v8h*)(bp);
    v8h bhi = *(const v8h*)(bp + 16);
    v16h bfrag = __builtin_shufflevector(blo, bhi,
        0,1,2,3,4,5,6,7,8,9,10,11,12,13,14,15);
    __builtin_prefetch(ap + 32, 0, 1);
    acc = __builtin_amdgcn_wmma_f32_16x16x32_f16(
        false, afrag, false, bfrag, (short)0, acc, false, false);
  }
  // C layout: VGPR v -> row m0 + half*8 + v, col n0 + l15
  float* crow = C + (long)(m0 + half * 8) * CH + n0 + l15;
#pragma unroll
  for (int v = 0; v < 8; ++v) crow[(long)v * CH] = acc[v];
}

// ---------------- conv1 attention ----------------
// a_src[n,h] = dot(h1[n, h*32 .. h*32+31], att_src[h]) ; same for dst
__global__ void attn1_kernel(const float* __restrict__ h1,
                             const float* __restrict__ att_s,
                             const float* __restrict__ att_d,
                             float* __restrict__ aS, float* __restrict__ aD) {
  int i = blockIdx.x * blockDim.x + threadIdx.x;    // i = n*4 + h
  if (i >= NN * NHEAD) return;
  int n = i >> 2, h = i & 3;
  const float* row = h1 + (long)n * CH + h * 32;
  const float* ws = att_s + h * 32;
  const float* wd = att_d + h * 32;
  float s = 0.f, d = 0.f;
#pragma unroll
  for (int c = 0; c < 32; ++c) { s += row[c] * ws[c]; d += row[c] * wd[c]; }
  aS[i] = s; aD[i] = d;
}

__device__ inline void edge_sd(int e, const int* ei, int& src, int& dst) {
  if (e < EE) { src = ei[e]; dst = ei[EE + e]; }
  else        { src = dst = e - EE; }
}

__global__ void edge_max1_kernel(const float* __restrict__ aS,
                                 const float* __restrict__ aD,
                                 const int* __restrict__ ei,
                                 float* __restrict__ el, float* __restrict__ m) {
  int e = blockIdx.x * blockDim.x + threadIdx.x;
  if (e >= ET) return;
  int src, dst; edge_sd(e, ei, src, dst);
#pragma unroll
  for (int h = 0; h < NHEAD; ++h) {
    float v = aS[src * NHEAD + h] + aD[dst * NHEAD + h];
    v = v > 0.f ? v : 0.2f * v;                     // leaky_relu
    el[(long)e * NHEAD + h] = v;
    atomicMaxF(&m[dst * NHEAD + h], v);
  }
}

__global__ void edge_expsum1_kernel(const float* __restrict__ el,
                                    const float* __restrict__ m,
                                    const int* __restrict__ ei,
                                    float* __restrict__ s) {
  int e = blockIdx.x * blockDim.x + threadIdx.x;
  if (e >= ET) return;
  int src, dst; edge_sd(e, ei, src, dst);
#pragma unroll
  for (int h = 0; h < NHEAD; ++h) {
    float ex = expf(el[(long)e * NHEAD + h] - m[dst * NHEAD + h]);
    atomicAdd(&s[dst * NHEAD + h], ex);
  }
}

// warp per edge; lane handles channels lane, lane+32, lane+64, lane+96
__global__ void edge_scatter1_kernel(const float* __restrict__ h1,
                                     const float* __restrict__ el,
                                     const float* __restrict__ m,
                                     const float* __restrict__ s,
                                     const int* __restrict__ ei,
                                     float* __restrict__ acc) {
  int e    = blockIdx.x * (blockDim.x >> 5) + (threadIdx.x >> 5);
  int lane = threadIdx.x & 31;
  if (e >= ET) return;
  int src, dst; edge_sd(e, ei, src, dst);
  float alpha[NHEAD];
#pragma unroll
  for (int h = 0; h < NHEAD; ++h) {
    float ex = expf(el[(long)e * NHEAD + h] - m[dst * NHEAD + h]);
    alpha[h] = ex / (s[dst * NHEAD + h] + 1e-16f);
  }
  const float* hr = h1 + (long)src * CH;
  float* orow = acc + (long)dst * CH;
#pragma unroll
  for (int j = 0; j < 4; ++j) {
    int c = lane + 32 * j;                          // head = j
    atomicAdd(&orow[c], hr[c] * alpha[j]);
  }
}

// ---------------- BatchNorm stats + apply (fused bias/ELU/residual/f16) ----------------
__global__ void bn_stats_kernel(const float* __restrict__ acc,
                                const float* __restrict__ b1,
                                float* __restrict__ sums) {
  int c = threadIdx.x;                              // 128 threads
  float s = 0.f, s2 = 0.f;
  for (int n = blockIdx.x; n < NN; n += gridDim.x) {
    float v = acc[(long)n * CH + c] + b1[c];
    s += v; s2 += v * v;
  }
  atomicAdd(&sums[c], s);
  atomicAdd(&sums[CH + c], s2);
}

__global__ void bn_apply_kernel(const float* __restrict__ acc,
                                const float* __restrict__ b1,
                                const float* __restrict__ sums,
                                const float* __restrict__ gamma,
                                const float* __restrict__ beta,
                                const float* __restrict__ x,
                                _Float16* __restrict__ y16) {
  int i = blockIdx.x * blockDim.x + threadIdx.x;
  if (i >= NN * CH) return;
  int c = i & 127;
  float v   = acc[i] + b1[c];
  float mu  = sums[c] * (1.f / NN);
  float var = sums[CH + c] * (1.f / NN) - mu * mu;
  float yh  = (v - mu) * rsqrtf(var + 1e-5f) * gamma[c] + beta[c];
  float ye  = yh > 0.f ? yh : (expf(yh) - 1.f);     // ELU
  y16[i] = (_Float16)(ye + x[i]);                   // residual
}

// ---------------- conv2 attention (H=1, C=128) ----------------
__global__ void attn2_kernel(const float* __restrict__ h2,
                             const float* __restrict__ att_s,
                             const float* __restrict__ att_d,
                             float* __restrict__ aS, float* __restrict__ aD) {
  int n = blockIdx.x * blockDim.x + threadIdx.x;
  if (n >= NN) return;
  const float* row = h2 + (long)n * CH;
  float s = 0.f, d = 0.f;
  for (int c = 0; c < CH; ++c) { s += row[c] * att_s[c]; d += row[c] * att_d[c]; }
  aS[n] = s; aD[n] = d;
}

__global__ void edge_max2_kernel(const float* __restrict__ aS,
                                 const float* __restrict__ aD,
                                 const int* __restrict__ ei,
                                 float* __restrict__ el, float* __restrict__ m) {
  int e = blockIdx.x * blockDim.x + threadIdx.x;
  if (e >= ET) return;
  int src, dst; edge_sd(e, ei, src, dst);
  float v = aS[src] + aD[dst];
  v = v > 0.f ? v : 0.2f * v;
  el[e] = v;
  atomicMaxF(&m[dst], v);
}

__global__ void edge_expsum2_kernel(const float* __restrict__ el,
                                    const float* __restrict__ m,
                                    const int* __restrict__ ei,
                                    float* __restrict__ s) {
  int e = blockIdx.x * blockDim.x + threadIdx.x;
  if (e >= ET) return;
  int src, dst; edge_sd(e, ei, src, dst);
  atomicAdd(&s[dst], expf(el[e] - m[dst]));
}

__global__ void edge_scatter2_kernel(const float* __restrict__ h2,
                                     const float* __restrict__ el,
                                     const float* __restrict__ m,
                                     const float* __restrict__ s,
                                     const int* __restrict__ ei,
                                     float* __restrict__ out) {
  int e    = blockIdx.x * (blockDim.x >> 5) + (threadIdx.x >> 5);
  int lane = threadIdx.x & 31;
  if (e >= ET) return;
  int src, dst; edge_sd(e, ei, src, dst);
  float alpha = expf(el[e] - m[dst]) / (s[dst] + 1e-16f);
  const float* hr = h2 + (long)src * CH;
  float* orow = out + (long)dst * CH;
#pragma unroll
  for (int j = 0; j < 4; ++j) {
    int c = lane + 32 * j;
    atomicAdd(&orow[c], hr[c] * alpha);
  }
}

__global__ void bias2_kernel(float* __restrict__ out,
                             const float* __restrict__ b2) {
  int i = blockIdx.x * blockDim.x + threadIdx.x;
  if (i >= NN * CH) return;
  out[i] += b2[i & 127];
}

// ---------------- host launcher ----------------
extern "C" void kernel_launch(void* const* d_in, const int* in_sizes, int n_in,
                              void* d_out, int out_size, void* d_ws, size_t ws_size,
                              hipStream_t stream) {
  const float* x        = (const float*)d_in[0];
  const int*   ei       = (const int*)  d_in[1];
  const float* W1       = (const float*)d_in[2];
  const float* att_src1 = (const float*)d_in[3];
  const float* att_dst1 = (const float*)d_in[4];
  const float* b1       = (const float*)d_in[5];
  const float* W2       = (const float*)d_in[6];
  const float* att_src2 = (const float*)d_in[7];
  const float* att_dst2 = (const float*)d_in[8];
  const float* b2       = (const float*)d_in[9];
  const float* gamma    = (const float*)d_in[10];
  const float* beta     = (const float*)d_in[11];
  float* out = (float*)d_out;

  // workspace carve-up (256B aligned)
  char* base = (char*)d_ws;
  size_t o = 0;
  auto carve = [&](size_t bytes) -> char* {
    char* p = base + o;
    o += (bytes + 255) & ~(size_t)255;
    return p;
  };
  _Float16* x16  = (_Float16*)carve((size_t)NN * CH * 2);   // reused as y16
  _Float16* w1t  = (_Float16*)carve((size_t)CH * CH * 2);
  _Float16* w2t  = (_Float16*)carve((size_t)CH * CH * 2);
  float* h1      = (float*)carve((size_t)NN * CH * 4);      // reused as h2
  float* acc1    = (float*)carve((size_t)NN * CH * 4);
  float* aS1     = (float*)carve((size_t)NN * NHEAD * 4);
  float* aD1     = (float*)carve((size_t)NN * NHEAD * 4);
  float* m1      = (float*)carve((size_t)NN * NHEAD * 4);
  float* s1      = (float*)carve((size_t)NN * NHEAD * 4);
  float* el1     = (float*)carve((size_t)ET * NHEAD * 4);   // reused as el2
  float* aS2     = (float*)carve((size_t)NN * 4);
  float* aD2     = (float*)carve((size_t)NN * 4);
  float* m2      = (float*)carve((size_t)NN * 4);
  float* s2      = (float*)carve((size_t)NN * 4);
  float* sums    = (float*)carve((size_t)2 * CH * 4);
  _Float16* y16  = x16;
  float* h2      = h1;
  float* el2     = el1;

  // deterministic re-init each call (graph-capture-safe async memsets)
  hipMemsetAsync(m1, 0xFF, (size_t)NN * NHEAD * 4, stream);  // -NaN sentinel (int/uint trick safe)
  hipMemsetAsync(s1, 0,    (size_t)NN * NHEAD * 4, stream);
  hipMemsetAsync(m2, 0xFF, (size_t)NN * 4, stream);
  hipMemsetAsync(s2, 0,    (size_t)NN * 4, stream);
  hipMemsetAsync(sums, 0,  (size_t)2 * CH * 4, stream);
  hipMemsetAsync(acc1, 0,  (size_t)NN * CH * 4, stream);
  hipMemsetAsync(out,  0,  (size_t)NN * CH * 4, stream);

  const int T = 256;
  const int nElem  = NN * CH;                  // 12.8M
  const int gElem  = (nElem + T - 1) / T;
  const int gEdge  = (ET + T - 1) / T;
  const int gWarpE = (ET + (T / 32) - 1) / (T / 32);
  const int gTiles = ((NN + 15) / 16) * (CH / 16) / 8 + 1;  // 8 waves/block

  // prep
  cvt_f32_to_f16_kernel<<<gElem, T, 0, stream>>>(x, x16, nElem);
  transpose_w_kernel<<<(CH * CH + T - 1) / T, T, 0, stream>>>(W1, w1t);
  transpose_w_kernel<<<(CH * CH + T - 1) / T, T, 0, stream>>>(W2, w2t);

  // conv1: h1 = x @ W1 (WMMA)
  wmma_gemm128_kernel<<<gTiles, T, 0, stream>>>(x16, w1t, h1, NN);
  attn1_kernel<<<(NN * NHEAD + T - 1) / T, T, 0, stream>>>(h1, att_src1, att_dst1, aS1, aD1);
  edge_max1_kernel<<<gEdge, T, 0, stream>>>(aS1, aD1, ei, el1, m1);
  edge_expsum1_kernel<<<gEdge, T, 0, stream>>>(el1, m1, ei, s1);
  edge_scatter1_kernel<<<gWarpE, T, 0, stream>>>(h1, el1, m1, s1, ei, acc1);

  // BN + ELU + residual -> y16
  bn_stats_kernel<<<512, CH, 0, stream>>>(acc1, b1, sums);
  bn_apply_kernel<<<gElem, T, 0, stream>>>(acc1, b1, sums, gamma, beta, x, y16);

  // conv2: h2 = y @ W2 (WMMA)
  wmma_gemm128_kernel<<<gTiles, T, 0, stream>>>(y16, w2t, h2, NN);
  attn2_kernel<<<(NN + T - 1) / T, T, 0, stream>>>(h2, att_src2, att_dst2, aS2, aD2);
  edge_max2_kernel<<<gEdge, T, 0, stream>>>(aS2, aD2, ei, el2, m2);
  edge_expsum2_kernel<<<gEdge, T, 0, stream>>>(el2, m2, ei, s2);
  edge_scatter2_kernel<<<gWarpE, T, 0, stream>>>(h2, el2, m2, s2, ei, out);
  bias2_kernel<<<gElem, T, 0, stream>>>(out, b2);
}
